// QuantBasicBlock_87016037417496
// MI455X (gfx1250) — matrix operand order repaired and checked by reading, not compiled, not run on hardware
//
#include <hip/hip_runtime.h>
#include <hip/hip_bf16.h>

// ---------------------------------------------------------------------------
// QuantBasicBlock for MI455X (gfx1250): conv3x3 as implicit GEMM on WMMA.
//   A (16x32 f16)  = quantized weights, pre-packed to WMMA A-fragment layout,
//                    staged per-block in LDS (ds_load_b128 in the hot loop)
//   B (32x16 f16)  = halo-padded NHWC f16 activations -> branchless loads at
//                    immediate offsets from one base address
//   D (16x16 f32)  = 16 out-channels x 16 pixels, accumulated over 9 taps
// ---------------------------------------------------------------------------

typedef __attribute__((ext_vector_type(16))) _Float16 v16h;
typedef __attribute__((ext_vector_type(8)))  float    v8f;

#define NN 16
#define CC 32
#define HH 256
#define WW 256
#define HP 258                          // padded rows (1-pixel halo)
#define WP 258                          // padded cols
#define HWP (HH * WW)                   // 65536
#define PIX (NN * HWP)                  // 1,048,576 pixels
#define ELEMS ((size_t)PIX * CC)        // 33,554,432 elements
#define PAD_HALVES ((size_t)NN * HP * WP * CC)   // padded f16 activation elems
#define QMAXF 7.0f
#define BN_EPS 1e-5f

#define WPACK_PER_CONV (9 * 2 * 32 * 16)   // taps * mtiles * lanes * halves

// ---------------- weight quantization + A-fragment packing -----------------
// One wave per (conv, o, tap). lane = input channel i. Computes per-(o,tap)
// scale s = max_i|w|/7 (block quant, BLK=32 == I), quantizes, and scatters the
// f16 value into a 16x32 A-fragment layout with contiguous k = 16*(lane/16) + half.
__global__ void qbb_pack_weights(const float* __restrict__ w1,
                                 const float* __restrict__ w2,
                                 _Float16* __restrict__ wpack) {
    int gid  = (blockIdx.x * blockDim.x + threadIdx.x) >> 5;  // wave id
    int lane = threadIdx.x & 31;                              // input channel i
    if (gid >= 2 * 32 * 9) return;
    int conv = gid / (32 * 9);
    int rem  = gid - conv * (32 * 9);
    int o    = rem / 9;
    int t    = rem - o * 9;

    const float* w = conv ? w2 : w1;
    float v = w[(size_t)(o * 32 + lane) * 9 + t];

    float m = fabsf(v);
    #pragma unroll
    for (int d = 16; d; d >>= 1) m = fmaxf(m, __shfl_xor(m, d, 32));
    float s = fmaxf(m / QMAXF, 1e-12f);
    float q = fminf(fmaxf(rintf(v / s), -QMAXF), QMAXF);
    float wq = q * s;

    // inverse of A-fragment layout: which (frag-lane, half) holds (o, i)?
    int hi = lane >> 4;                       // i in {16..31} -> hi lane
    int j  = lane & 15;                       // half index within v16h
    int mt = o >> 4;
    int fl = (o & 15) + 16 * hi;
    wpack[(size_t)conv * WPACK_PER_CONV + ((t * 2 + mt) * 32 + fl) * 16 + j] =
        (_Float16)wq;
}

// ---------------- NCHW f32 -> halo-padded NHWC f16 (conv1 input) -----------
__global__ void qbb_to_nhwc_f16(const float* __restrict__ x,
                                _Float16* __restrict__ xh) {
    size_t tid = (size_t)blockIdx.x * blockDim.x + threadIdx.x;
    size_t pix = tid >> 5;
    int c = (int)(tid & 31);
    size_t n  = pix >> 16;          // / HWP
    size_t hw = pix & (HWP - 1);
    int h = (int)(hw >> 8);
    int w = (int)(hw & 255);
    xh[((n * HP + h + 1) * WP + (w + 1)) * CC + c] =
        (_Float16)x[(n * CC + c) * HWP + hw];
}

// ---------------- conv3x3 via WMMA implicit GEMM ---------------------------
// One wave per (n, h, 16-pixel tile). 18 v_wmma_f32_16x16x32_f16 per wave.
// Branch-free: halo padding makes every tap load unconditional, and all 9 tap
// addresses are immediate offsets from one base pointer.
__global__ void __launch_bounds__(256)
qbb_conv3x3_wmma(const _Float16* __restrict__ act,   // padded NHWC f16
                 const _Float16* __restrict__ wp,    // packed A fragments
                 float* __restrict__ outb) {         // dense NHWC f32
    __shared__ _Float16 sa[WPACK_PER_CONV];          // 18 KB of 320 KB LDS
    {
        const uint4* s = (const uint4*)wp;
        uint4* d = (uint4*)sa;
        for (int i = threadIdx.x; i < WPACK_PER_CONV / 8; i += 256) d[i] = s[i];
    }
    __syncthreads();

    int wgid = blockIdx.x * 8 + (threadIdx.x >> 5);
    int lane = threadIdx.x & 31;
    const int TW = WW / 16;
    int wt  = wgid % TW;
    int rem = wgid / TW;
    int h   = rem % HH;
    int n   = rem / HH;
    int pN    = lane & 15;   // pixel within tile (B/D matrix N index)
    int khalf = lane >> 4;   // which 16-channel half of K this lane carries

    // base = padded pixel (h, w0+pN) i.e. tap (dh=-1, dw=-1)
    const _Float16* brow =
        act + ((size_t)(n * HP + h) * WP + (wt * 16 + pN)) * CC + khalf * 16;

    // prefetch the next tile-row's activations
    if (h + 2 < HH)
        __builtin_prefetch(brow + (size_t)3 * WP * CC, 0, 1);

    v8f acc0 = {};   // out channels [khalf*8 .. khalf*8+7]
    v8f acc1 = {};   // out channels [16+khalf*8 .. 16+khalf*8+7]

    #pragma unroll
    for (int t = 0; t < 9; ++t) {
        const int off = ((t / 3) * WP + (t % 3)) * CC;   // compile-time imm
        v16h b  = *(const v16h*)(brow + off);
        v16h a0 = *(const v16h*)(sa + ((t * 2 + 0) * 32 + lane) * 16);
        v16h a1 = *(const v16h*)(sa + ((t * 2 + 1) * 32 + lane) * 16);
        acc0 = __builtin_amdgcn_wmma_f32_16x16x32_f16(false, a0, false, b,
                                                      (short)0, acc0, false, false);
        acc1 = __builtin_amdgcn_wmma_f32_16x16x32_f16(false, a1, false, b,
                                                      (short)0, acc1, false, false);
    }

    // D layout: VGPR r holds M=r (lanes 0-15) / M=r+8 (lanes 16-31), N=lane%16
    size_t pixi = (size_t)n * HWP + (size_t)h * WW + (wt * 16 + pN);
    float* base = outb + pixi * 32 + khalf * 8;   // 32B aligned
    *(v8f*)(base)      = acc0;                    // channels mt=0
    *(v8f*)(base + 16) = acc1;                    // channels mt=1
}

// ---------------- per-channel sum / sumsq (BN batch stats) -----------------
// convb (134 MB) fits in the 192 MB L2, so this pass is L2-resident.
__global__ void qbb_bn_stats(const float* __restrict__ buf,
                             float* __restrict__ stats) {  // [64]: sum, sumsq
    __shared__ float ls[64];
    if (threadIdx.x < 64) ls[threadIdx.x] = 0.0f;
    __syncthreads();
    int c = threadIdx.x & 31;   // stride is a multiple of 32 -> channel fixed
    float s = 0.0f, s2 = 0.0f;
    size_t stride = (size_t)gridDim.x * blockDim.x;
    for (size_t i = (size_t)blockIdx.x * blockDim.x + threadIdx.x; i < ELEMS;
         i += stride) {
        float v = buf[i];
        s += v;
        s2 += v * v;
    }
    atomicAdd(&ls[c], s);         // ds_add_f32
    atomicAdd(&ls[c + 32], s2);
    __syncthreads();
    if (threadIdx.x < 64) atomicAdd(&stats[threadIdx.x], ls[threadIdx.x]);
}

__device__ __forceinline__ float qbb_bfp(float v) {
    // v >= 0 (post-ReLU). Shared exponent over the wave's 32 channels.
    float m = v;
    #pragma unroll
    for (int d = 16; d; d >>= 1) m = fmaxf(m, __shfl_xor(m, d, 32));
    m = fmaxf(m, 1e-12f);
    int e = (int)((__float_as_uint(m) >> 23) & 0xFF) - 127;   // floor(log2 m)
    float lsb = __uint_as_float((unsigned)(e + 125) << 23);   // 2^(e-2)
    float q = fminf(rintf(v / lsb), QMAXF);
    return q * lsb;
}

// ---------------- BN1 + ReLU + BFP -> padded f16 NHWC (conv2 input) --------
__global__ void qbb_bn_relu_bfp_mid(const float* __restrict__ convb,
                                    const float* __restrict__ stats,
                                    const float* __restrict__ gamma,
                                    const float* __restrict__ beta,
                                    _Float16* __restrict__ act1) {
    size_t tid = (size_t)blockIdx.x * blockDim.x + threadIdx.x;
    int c = threadIdx.x & 31;
    float mu  = stats[c] * (1.0f / PIX);
    float var = stats[c + 32] * (1.0f / PIX) - mu * mu;
    float sc  = gamma[c] * rsqrtf(var + BN_EPS);
    float sh  = beta[c] - mu * sc;
    float v = fmaxf(convb[tid] * sc + sh, 0.0f);

    size_t pix = tid >> 5;
    size_t n  = pix >> 16;
    size_t hw = pix & (HWP - 1);
    int h = (int)(hw >> 8);
    int w = (int)(hw & 255);
    act1[((n * HP + h + 1) * WP + (w + 1)) * CC + c] = (_Float16)qbb_bfp(v);
}

// ---------------- BN2 + shortcut + ReLU + BFP -> f32 NCHW output -----------
__global__ void qbb_bn_add_relu_bfp_out(const float* __restrict__ convb,
                                        const float* __restrict__ stats,
                                        const float* __restrict__ gamma,
                                        const float* __restrict__ beta,
                                        const float* __restrict__ x,
                                        float* __restrict__ out) {
    size_t tid = (size_t)blockIdx.x * blockDim.x + threadIdx.x;
    int c = threadIdx.x & 31;
    size_t pix = tid >> 5;
    size_t n  = pix >> 16;
    size_t hw = pix & (HWP - 1);
    size_t nchw = (n * CC + c) * HWP + hw;

    float mu  = stats[c] * (1.0f / PIX);
    float var = stats[c + 32] * (1.0f / PIX) - mu * mu;
    float sc  = gamma[c] * rsqrtf(var + BN_EPS);
    float sh  = beta[c] - mu * sc;

    float v = fmaxf(convb[tid] * sc + sh + x[nchw], 0.0f);
    out[nchw] = qbb_bfp(v);
}

// ---------------------------------------------------------------------------
extern "C" void kernel_launch(void* const* d_in, const int* in_sizes, int n_in,
                              void* d_out, int out_size, void* d_ws, size_t ws_size,
                              hipStream_t stream) {
    (void)in_sizes; (void)n_in; (void)out_size; (void)ws_size;
    const float* x      = (const float*)d_in[0];
    const float* w1     = (const float*)d_in[1];
    const float* gamma1 = (const float*)d_in[2];
    const float* beta1  = (const float*)d_in[3];
    const float* w2     = (const float*)d_in[4];
    const float* gamma2 = (const float*)d_in[5];
    const float* beta2  = (const float*)d_in[6];
    float* out = (float*)d_out;

    const size_t PAD_BYTES = PAD_HALVES * sizeof(_Float16);  // 68,161,536 B
    char* ws = (char*)d_ws;
    float*    convb  = (float*)(ws);                                  // 134,217,728 B
    _Float16* xh     = (_Float16*)(ws + 134217728);                   // padded f16
    _Float16* act1   = (_Float16*)(ws + 134217728 + PAD_BYTES);       // padded f16
    char*     tail   = ws + 134217728 + 2 * PAD_BYTES;
    _Float16* wpack  = (_Float16*)(tail);                             // 36,864 B
    float*    stats1 = (float*)(tail + 36864);                        // 256 B
    float*    stats2 = (float*)(tail + 36864 + 256);                  // 256 B

    // zero halos (and interiors; interiors are fully overwritten) + stats
    hipMemsetAsync(xh,   0, PAD_BYTES, stream);
    hipMemsetAsync(act1, 0, PAD_BYTES, stream);
    hipMemsetAsync(stats1, 0, 64 * sizeof(float), stream);
    hipMemsetAsync(stats2, 0, 64 * sizeof(float), stream);

    // weight quant + pack: 2*32*9 = 576 waves
    qbb_pack_weights<<<72, 256, 0, stream>>>(w1, w2, wpack);

    // x -> padded NHWC f16
    qbb_to_nhwc_f16<<<(unsigned)(ELEMS / 256), 256, 0, stream>>>(x, xh);

    // conv1 (WMMA): 65536 tile-waves / 8 waves-per-block
    qbb_conv3x3_wmma<<<8192, 256, 0, stream>>>(xh, wpack, convb);
    qbb_bn_stats<<<2048, 256, 0, stream>>>(convb, stats1);
    qbb_bn_relu_bfp_mid<<<(unsigned)(ELEMS / 256), 256, 0, stream>>>(
        convb, stats1, gamma1, beta1, act1);

    // conv2 (WMMA) + BN2 + shortcut + ReLU + BFP
    qbb_conv3x3_wmma<<<8192, 256, 0, stream>>>(act1, wpack + WPACK_PER_CONV, convb);
    qbb_bn_stats<<<2048, 256, 0, stream>>>(convb, stats2);
    qbb_bn_add_relu_bfp_out<<<(unsigned)(ELEMS / 256), 256, 0, stream>>>(
        convb, stats2, gamma2, beta2, x, out);
}
